// BlockAttention_31275951849957
// MI455X (gfx1250) — matrix operand order repaired and checked by reading, not compiled, not run on hardware
//
#include <hip/hip_runtime.h>

// Problem constants (from reference): B=2, S=4096, E=1024, H=16, D=64, block=256
#define E_    1024
#define S_    4096
#define B_    2
#define H_    16
#define D_    64
#define BLK_  256
#define NB_   (S_ / BLK_)       // 16
#define ROWS_ (B_ * S_)         // 8192

typedef __bf16 bf16_t;
typedef __attribute__((ext_vector_type(16))) __bf16 v16bf;
typedef __attribute__((ext_vector_type(8)))  __bf16 v8bf;
typedef __attribute__((ext_vector_type(8)))  float  v8f;

union ABfrag { v16bf v; v8bf h[2]; };

__device__ __forceinline__ v8f vzero() {
  v8f z = {0.f, 0.f, 0.f, 0.f, 0.f, 0.f, 0.f, 0.f};
  return z;
}

__device__ __forceinline__ v8f wmma_bf16(const ABfrag& a, const ABfrag& b, v8f c) {
  // D = A(16x32 bf16) * B(32x16 bf16) + C(16x16 f32)
  return __builtin_amdgcn_wmma_f32_16x16x32_bf16(false, a.v, false, b.v,
                                                 (short)0, c, false, false);
}

// two f32 -> packed bf16 pair (lowers to v_cvt_pk_bf16_f32)
__device__ __forceinline__ unsigned pack_bf16(float a, float b) {
  union { bf16_t h[2]; unsigned u; } p;
  p.h[0] = (bf16_t)a;
  p.h[1] = (bf16_t)b;
  return p.u;
}

// CDNA5 async global->LDS copy, 16B per lane, tracked on ASYNCcnt.
// lds_addr = low 32 bits of a __shared__ pointer (LDS aperture offset).
__device__ __forceinline__ void async_load_b128(unsigned lds_addr, const void* gptr) {
  asm volatile("global_load_async_to_lds_b128 %0, %1, off"
               :: "v"(lds_addr), "v"((unsigned long long)(uintptr_t)gptr)
               : "memory");
}
__device__ __forceinline__ void wait_async0() {
  asm volatile("s_wait_asynccnt 0x0" ::: "memory");
}

// ---------------------------------------------------------------------------
// Kernel 1: fused QKVG projection. grid = (ROWS/256, 4*16). blockIdx.y selects
// weight matrix m = y/16 and head strip h = y%16 (64 columns). Each of 8 waves
// computes a 32x64 f32 tile (two 16-row groups sharing every B fragment),
// K-loop over E in steps of 32 through bf16 LDS tiles staged with packed
// v_cvt_pk_bf16_f32 + b128/b32 stores. Epilogue: RoPE for q/k (the rotate-half
// partner of column d lives in accumulator tile j+-2 at the same lane/element),
// sigmoid for gate. q/k/v stored [B,nb,H,256,64] bf16, gate [row,E] bf16.
// ---------------------------------------------------------------------------
__global__ __launch_bounds__(256) void k_proj(
    const float* __restrict__ x,
    const float* __restrict__ Wq, const float* __restrict__ Wk,
    const float* __restrict__ Wv, const float* __restrict__ Wg,
    const float* __restrict__ cosT, const float* __restrict__ sinT,
    bf16_t* __restrict__ qw, bf16_t* __restrict__ kw,
    bf16_t* __restrict__ vw, bf16_t* __restrict__ gw)
{
  __shared__ __align__(16) bf16_t Ax[256][40];   // 256 rows x 32 K (pad 40)
  __shared__ __align__(16) bf16_t Bw[64][32];    // [n][k] transposed weight tile

  const int tid  = threadIdx.x;
  const int lane = tid & 31;
  const int wv   = tid >> 5;
  const int r0   = blockIdx.x * 256;
  const int m    = blockIdx.y >> 4;       // 0=q 1=k 2=v 3=gate
  const int hh   = blockIdx.y & 15;       // head index == 64-col strip
  const float* W = (m == 0) ? Wq : (m == 1) ? Wk : (m == 2) ? Wv : Wg;
  const int colW0 = hh * 64;

  v8f acc[2][4] = {{vzero(), vzero(), vzero(), vzero()},
                   {vzero(), vzero(), vzero(), vzero()}};

  const int off1 = (lane < 16) ? 0 : 8;    // A-frag K group base
  const int offB = (lane < 16) ? 0 : 16;   // unused; B-frag also uses off1

  for (int kk = 0; kk < E_ / 32; ++kk) {
    // stage x tile (256x32 f32 -> bf16), packed: 8x float4 loads, 4x b128 stores
#pragma unroll
    for (int i = 0; i < 2; ++i) {
      const int rr = (tid >> 1) + i * 128;
      const int ch = (tid & 1) * 16;
      const float* xp = &x[(size_t)(r0 + rr) * E_ + kk * 32 + ch];
      const float4 f0 = ((const float4*)xp)[0];
      const float4 f1 = ((const float4*)xp)[1];
      const float4 f2 = ((const float4*)xp)[2];
      const float4 f3 = ((const float4*)xp)[3];
      uint4 u0, u1;
      u0.x = pack_bf16(f0.x, f0.y); u0.y = pack_bf16(f0.z, f0.w);
      u0.z = pack_bf16(f1.x, f1.y); u0.w = pack_bf16(f1.z, f1.w);
      u1.x = pack_bf16(f2.x, f2.y); u1.y = pack_bf16(f2.z, f2.w);
      u1.z = pack_bf16(f3.x, f3.y); u1.w = pack_bf16(f3.z, f3.w);
      *(uint4*)&Ax[rr][ch]     = u0;
      *(uint4*)&Ax[rr][ch + 8] = u1;
    }
    // stage weight tile transposed (32x64 -> [n][k]), packed k-pairs
#pragma unroll
    for (int i = 0; i < 4; ++i) {
      const int p  = tid + i * 256;
      const int nn = p & 63;
      const int kp = p >> 6;              // 0..15
      const float w0 = W[(size_t)(kk * 32 + 2 * kp)     * E_ + colW0 + nn];
      const float w1 = W[(size_t)(kk * 32 + 2 * kp + 1) * E_ + colW0 + nn];
      *(unsigned*)&Bw[nn][2 * kp] = pack_bf16(w0, w1);
    }
    if (kk + 1 < E_ / 32)  // hint the next x tile toward L2 (global_prefetch_b8)
      __builtin_prefetch(&x[(size_t)(r0 + tid) * E_ + (kk + 1) * 32], 0, 1);
    __syncthreads();

    ABfrag a[2];
#pragma unroll
    for (int g2 = 0; g2 < 2; ++g2) {
      const int row = wv * 32 + g2 * 16 + (lane & 15);
      a[g2].h[0] = *(const v8bf*)&Ax[row][off1];
      a[g2].h[1] = *(const v8bf*)&Ax[row][off1 + 16];
    }
#pragma unroll
    for (int j = 0; j < 4; ++j) {
      ABfrag b;
      const bf16_t* bp = &Bw[j * 16 + (lane & 15)][0];
      b.h[0] = *(const v8bf*)&bp[off1];
      b.h[1] = *(const v8bf*)&bp[off1 + 16];
      acc[0][j] = wmma_bf16(a[0], b, acc[0][j]);
      acc[1][j] = wmma_bf16(a[1], b, acc[1][j]);
    }
    __syncthreads();
  }

  // epilogue
#pragma unroll
  for (int g2 = 0; g2 < 2; ++g2) {
#pragma unroll
    for (int j = 0; j < 4; ++j) {
      const int d = j * 16 + (lane & 15);
#pragma unroll
      for (int r = 0; r < 8; ++r) {
        const int mrow = r + ((lane < 16) ? 0 : 8);
        const int rg = r0 + wv * 32 + g2 * 16 + mrow;   // global row
        const int bb = rg / S_;
        const int ss = rg - bb * S_;
        float val = acc[g2][j][r];
        if (m < 2) {
          // RoPE: partner of column d is d+-32 -> tile j+-2, same lane/elem
          const float partner = (j < 2) ? -acc[g2][j + 2][r] : acc[g2][j - 2][r];
          const float cv = cosT[ss * D_ + d];
          const float sv = sinT[ss * D_ + d];
          val = val * cv + partner * sv;
          const int nblk = ss >> 8, rb = ss & 255;
          const size_t dst = ((((size_t)bb * NB_ + nblk) * H_ + hh) * BLK_ + rb) * D_ + d;
          (m == 0 ? qw : kw)[dst] = (bf16_t)val;
        } else if (m == 2) {
          const int nblk = ss >> 8, rb = ss & 255;
          const size_t dst = ((((size_t)bb * NB_ + nblk) * H_ + hh) * BLK_ + rb) * D_ + d;
          vw[dst] = (bf16_t)val;
        } else {
          const float sg = 1.f / (1.f + __expf(-val));
          gw[(size_t)rg * E_ + hh * 64 + d] = (bf16_t)sg;
        }
      }
    }
  }
}

// ---------------------------------------------------------------------------
// Kernel 2: block-diagonal attention. One workgroup per (b, block, head).
// K staged in LDS [key][d] via CDNA5 async global->LDS b128 copies (ASYNCcnt);
// V staged transposed [d][key] with packed b32 stores so both WMMA B-fragments
// are contiguous LDS reads. 8 waves x 32 q-rows (two 16-row groups). Scores
// (16x256) kept in registers, softmax via half-wave __shfl_xor reductions,
// probs re-laid-out through a per-wave LDS buffer into A-fragment order, then
// O = P*V, fused sigmoid-gate, bf16 store.
// ---------------------------------------------------------------------------
__global__ __launch_bounds__(256) void k_attn(
    const bf16_t* __restrict__ qw, const bf16_t* __restrict__ kw,
    const bf16_t* __restrict__ vw, const bf16_t* __restrict__ gw,
    bf16_t* __restrict__ attw)
{
  __shared__ __align__(16) bf16_t sK[256][72];        // [key][d]
  __shared__ __align__(16) bf16_t sVt[64][264];       // [d][key]
  __shared__ __align__(16) bf16_t sP[8][16 * 264];    // per-wave prob buffer

  const int tid = threadIdx.x, lane = tid & 31, wv = tid >> 5;
  const int bid  = blockIdx.x;
  const int h    = bid & 15;
  const int nblk = (bid >> 4) & 15;
  const int bb   = bid >> 8;
  const size_t base = (((size_t)bb * NB_ + nblk) * H_ + h) * (size_t)(BLK_ * D_);
  const bf16_t* Q = qw + base;
  const bf16_t* K = kw + base;
  const bf16_t* V = vw + base;

  // stage K: one 64-elem row per thread, async global->LDS (8 x 16B per lane)
  {
    const int rowk = tid;
#pragma unroll
    for (int i = 0; i < 8; ++i)
      async_load_b128((unsigned)(uintptr_t)&sK[rowk][i * 8],
                      &K[rowk * 64 + i * 8]);
  }
  // stage V transposed, packed key-pairs
  {
    const int d  = tid & 63;
    const int k0 = (tid >> 6) * 64;
    for (int key = k0; key < k0 + 64; key += 2) {
      union { bf16_t h[2]; unsigned u; } p;
      p.h[0] = V[key * 64 + d];
      p.h[1] = V[(key + 1) * 64 + d];
      *(unsigned*)&sVt[d][key] = p.u;
    }
  }
  wait_async0();
  __syncthreads();

  bf16_t* prW = &sP[wv][0];
  const int off1 = (lane < 16) ? 0 : 8;
  const int offB = (lane < 16) ? 0 : 16;
  const float scale = 0.125f;   // 1/sqrt(64)

  for (int g = 0; g < 2; ++g) {
    const int qrow0 = wv * 32 + g * 16;

    // Q A-fragments for both K-steps (d 0..31 and 32..63), straight from global
    ABfrag a0, a1;
    {
      const bf16_t* qp = Q + (qrow0 + (lane & 15)) * 64;
      a0.h[0] = *(const v8bf*)&qp[off1];
      a0.h[1] = *(const v8bf*)&qp[off1 + 16];
      a1.h[0] = *(const v8bf*)&qp[32 + off1];
      a1.h[1] = *(const v8bf*)&qp[32 + off1 + 16];
    }

    // scores: 16 key-tiles of 16, contraction over D=64 in two WMMAs
    v8f sc[16];
#pragma unroll
    for (int t = 0; t < 16; ++t) {
      v8f s = vzero();
      const bf16_t* kp = &sK[t * 16 + (lane & 15)][0];
      ABfrag b0, b1;
      b0.h[0] = *(const v8bf*)&kp[off1];
      b0.h[1] = *(const v8bf*)&kp[off1 + 16];
      b1.h[0] = *(const v8bf*)&kp[32 + off1];
      b1.h[1] = *(const v8bf*)&kp[32 + off1 + 16];
      s = wmma_bf16(a0, b0, s);
      s = wmma_bf16(a1, b1, s);
      sc[t] = s;
    }

    // softmax over 256 keys per row (rows live in 16-lane halves)
    v8f mx = sc[0];
#pragma unroll
    for (int t = 1; t < 16; ++t)
#pragma unroll
      for (int r = 0; r < 8; ++r) mx[r] = fmaxf(mx[r], sc[t][r]);
#pragma unroll
    for (int msk = 1; msk < 16; msk <<= 1)
#pragma unroll
      for (int r = 0; r < 8; ++r) mx[r] = fmaxf(mx[r], __shfl_xor(mx[r], msk, 32));

    v8f sum = vzero();
#pragma unroll
    for (int t = 0; t < 16; ++t)
#pragma unroll
      for (int r = 0; r < 8; ++r) {
        const float e = __expf((sc[t][r] - mx[r]) * scale);
        sc[t][r] = e;
        sum[r] += e;
      }
#pragma unroll
    for (int msk = 1; msk < 16; msk <<= 1)
#pragma unroll
      for (int r = 0; r < 8; ++r) sum[r] += __shfl_xor(sum[r], msk, 32);

    v8f inv;
#pragma unroll
    for (int r = 0; r < 8; ++r) inv[r] = 1.f / sum[r];

    // re-layout probs C->A through per-wave LDS buffer (packed bf16 pairs are
    // not possible here: adjacent keys live in different lanes) 
#pragma unroll
    for (int t = 0; t < 16; ++t)
#pragma unroll
      for (int r = 0; r < 8; ++r) {
        const int mrow = r + ((lane < 16) ? 0 : 8);
        prW[mrow * 264 + t * 16 + (lane & 15)] = (bf16_t)(sc[t][r] * inv[r]);
      }

    // O = P(16x256) * V(256x64): 4 N-tiles x 8 K-steps
#pragma unroll
    for (int nt = 0; nt < 4; ++nt) {
      v8f o = vzero();
#pragma unroll
      for (int st = 0; st < 8; ++st) {
        ABfrag a;
        const bf16_t* pp = &prW[(lane & 15) * 264 + st * 32];
        a.h[0] = *(const v8bf*)&pp[off1];
        a.h[1] = *(const v8bf*)&pp[off1 + 16];
        ABfrag b;
        const bf16_t* vp = &sVt[nt * 16 + (lane & 15)][st * 32];
        b.h[0] = *(const v8bf*)&vp[off1];
        b.h[1] = *(const v8bf*)&vp[off1 + 16];
        o = wmma_bf16(a, b, o);
      }
      // fused sigmoid(gate) multiply + store
#pragma unroll
      for (int r = 0; r < 8; ++r) {
        const int mrow = r + ((lane < 16) ? 0 : 8);
        const int qr = qrow0 + mrow;
        const int d = nt * 16 + (lane & 15);
        const size_t rg = (size_t)bb * S_ + nblk * BLK_ + qr;
        const float gv = (float)gw[rg * E_ + h * 64 + d];
        attw[rg * E_ + h * 64 + d] = (bf16_t)(o[r] * gv);
      }
    }
  }
}

// ---------------------------------------------------------------------------
// Kernel 3: output projection att(bf16) @ Wo(f32->bf16) -> f32 out.
// Same 256-row tiling as k_proj; A tile is a straight bf16 b128 copy.
// grid = (ROWS/256, E/64).
// ---------------------------------------------------------------------------
__global__ __launch_bounds__(256) void k_out(
    const bf16_t* __restrict__ att, const float* __restrict__ Wo,
    float* __restrict__ out)
{
  __shared__ __align__(16) bf16_t Ax[256][40];
  __shared__ __align__(16) bf16_t Bw[64][32];

  const int tid = threadIdx.x, lane = tid & 31, wv = tid >> 5;
  const int r0 = blockIdx.x * 256;
  const int colW0 = blockIdx.y * 64;
  const int off1 = (lane < 16) ? 0 : 8;
  const int offB = (lane < 16) ? 0 : 16;

  v8f acc[2][4] = {{vzero(), vzero(), vzero(), vzero()},
                   {vzero(), vzero(), vzero(), vzero()}};

  for (int kk = 0; kk < E_ / 32; ++kk) {
#pragma unroll
    for (int i = 0; i < 2; ++i) {
      const int rr = (tid >> 1) + i * 128;
      const int ch = (tid & 1) * 16;
      const bf16_t* ap = &att[(size_t)(r0 + rr) * E_ + kk * 32 + ch];
      *(uint4*)&Ax[rr][ch]     = ((const uint4*)ap)[0];
      *(uint4*)&Ax[rr][ch + 8] = ((const uint4*)ap)[1];
    }
#pragma unroll
    for (int i = 0; i < 4; ++i) {
      const int p  = tid + i * 256;
      const int nn = p & 63;
      const int kp = p >> 6;
      const float w0 = Wo[(size_t)(kk * 32 + 2 * kp)     * E_ + colW0 + nn];
      const float w1 = Wo[(size_t)(kk * 32 + 2 * kp + 1) * E_ + colW0 + nn];
      *(unsigned*)&Bw[nn][2 * kp] = pack_bf16(w0, w1);
    }
    __syncthreads();

    ABfrag a[2];
#pragma unroll
    for (int g2 = 0; g2 < 2; ++g2) {
      const int row = wv * 32 + g2 * 16 + (lane & 15);
      a[g2].h[0] = *(const v8bf*)&Ax[row][off1];
      a[g2].h[1] = *(const v8bf*)&Ax[row][off1 + 16];
    }
#pragma unroll
    for (int j = 0; j < 4; ++j) {
      ABfrag b;
      const bf16_t* bp = &Bw[j * 16 + (lane & 15)][0];
      b.h[0] = *(const v8bf*)&bp[off1];
      b.h[1] = *(const v8bf*)&bp[off1 + 16];
      acc[0][j] = wmma_bf16(a[0], b, acc[0][j]);
      acc[1][j] = wmma_bf16(a[1], b, acc[1][j]);
    }
    __syncthreads();
  }

#pragma unroll
  for (int g2 = 0; g2 < 2; ++g2) {
#pragma unroll
    for (int j = 0; j < 4; ++j) {
#pragma unroll
      for (int r = 0; r < 8; ++r) {
        const int mrow = r + ((lane < 16) ? 0 : 8);
        const size_t rg = (size_t)(r0 + wv * 32 + g2 * 16 + mrow);
        out[rg * E_ + colW0 + j * 16 + (lane & 15)] = acc[g2][j][r];
      }
    }
  }
}

extern "C" void kernel_launch(void* const* d_in, const int* in_sizes, int n_in,
                              void* d_out, int out_size, void* d_ws, size_t ws_size,
                              hipStream_t stream) {
  (void)in_sizes; (void)n_in; (void)out_size; (void)ws_size;
  const float* x  = (const float*)d_in[0];
  const float* Wq = (const float*)d_in[1];
  const float* Wk = (const float*)d_in[2];
  const float* Wv = (const float*)d_in[3];
  const float* Wg = (const float*)d_in[4];
  const float* Wo = (const float*)d_in[5];
  const float* cs = (const float*)d_in[6];
  const float* sn = (const float*)d_in[7];
  // d_in[8] = block_size (256), constant-folded into the kernels.

  const size_t SZ = (size_t)ROWS_ * E_;      // 8Mi elements per buffer
  bf16_t* qw = (bf16_t*)d_ws;                // [B,nb,H,256,64]
  bf16_t* kw = qw + SZ;
  bf16_t* vw = kw + SZ;
  bf16_t* gw = vw + SZ;                      // sigmoid(gate), [row, E]
  bf16_t* aw = gw + SZ;                      // gated attention out, [row, E]

  k_proj<<<dim3(ROWS_ / 256, 64), dim3(256), 0, stream>>>(
      x, Wq, Wk, Wv, Wg, cs, sn, qw, kw, vw, gw);
  k_attn<<<dim3(B_ * NB_ * H_), dim3(256), 0, stream>>>(qw, kw, vw, gw, aw);
  k_out<<<dim3(ROWS_ / 256, E_ / 64), dim3(256), 0, stream>>>(
      aw, Wo, (float*)d_out);
}
